// MoE_18837726560674
// MI455X (gfx1250) — matrix operand order repaired and checked by reading, not compiled, not run on hardware
//
#include <hip/hip_runtime.h>
#include <hip/hip_bf16.h>
#include <math.h>

#define NN      10000
#define DEG     32
#define EE      (NN * DEG)          // 320000
#define IND     128
#define HID     256
#define NEXP    4
#define FANIN   (2 * IND + 1)       // 257

typedef _Float16 v16h __attribute__((ext_vector_type(16)));
typedef _Float16 v8h  __attribute__((ext_vector_type(8)));
typedef float    v8f  __attribute__((ext_vector_type(8)));

// ---- workspace layout (bytes) ----
#define Z_OFF      0u                       // NEXP*EE floats  = 5,120,000 B
#define GATES_OFF  5120000u                 // NN*4 floats     =   160,000 B
#define SUMSQ_OFF  5280000u                 // 4 floats
#define IMP_OFF    5280016u                 // 4 floats
#define LOAD_OFF   5280032u                 // 4 floats
#define RNORM_OFF  5280048u                 // 4 floats
#define XH_OFF     5280064u                 // NN*IND halves   = 2,560,000 B
#define W1H_OFF    7840064u                 // NEXP*16*8*32*16 halves = 524,288 B
// total ~8.37 MB

// ---------------------------------------------------------------------------
__global__ void k_cvt_x(const float* __restrict__ x, _Float16* __restrict__ xh, int n) {
    int i = blockIdx.x * blockDim.x + threadIdx.x;
    if (i < n) xh[i] = (_Float16)x[i];
}

// Pre-swizzle W1 (first 256 K-rows) into B-fragment-major f16 layout:
// dst[(((e*16 + ht)*8 + ks)*32 + lane)*16 + h] = W1[e][ks*32 + (lane>>4)*8 + h + (h&8)][ht*16 + (lane&15)]
// One thread per (e,ht,ks,lane) -> writes 16 contiguous halves (32 B).
__global__ void k_cvt_w1(const float* __restrict__ W1, _Float16* __restrict__ W1h) {
    int t = blockIdx.x * blockDim.x + threadIdx.x;     // 0 .. 16383
    if (t >= NEXP * 16 * 8 * 32) return;
    int lane = t & 31;
    int ks   = (t >> 5) & 7;
    int ht   = (t >> 8) & 15;
    int e    = t >> 12;
    int n     = lane & 15;
    int kbase = ks * 32 + (lane >> 4) * 8;
    const float* src = W1 + e * FANIN * HID + kbase * HID + ht * 16 + n;
    _Float16* dst = W1h + (size_t)t * 16;
    #pragma unroll
    for (int h = 0; h < 16; h++) dst[h] = (_Float16)src[(h + (h & 8)) * HID];
}

__global__ void k_zero(float* sumsq, float* imp, float* loadv) {
    int i = threadIdx.x;
    if (i < 4) { sumsq[i] = 0.f; imp[i] = 0.f; loadv[i] = 0.f; }
}

// ---------------------------------------------------------------------------
// Gating: one wave (32 lanes) per node. logits = x[n] @ w_gate; top-2 softmax.
__global__ void k_gating(const float* __restrict__ x, const float* __restrict__ wg,
                         float* __restrict__ gates, float* imp, float* loadv) {
    int wave = (blockIdx.x * blockDim.x + threadIdx.x) >> 5;
    int lane = threadIdx.x & 31;
    if (wave >= NN) return;
    const float* xr = x + wave * IND;
    float p0 = 0.f, p1 = 0.f, p2 = 0.f, p3 = 0.f;
    #pragma unroll
    for (int f = lane; f < IND; f += 32) {
        float xv = xr[f];
        const float* w = wg + f * NEXP;
        p0 += xv * w[0]; p1 += xv * w[1]; p2 += xv * w[2]; p3 += xv * w[3];
    }
    #pragma unroll
    for (int m = 16; m >= 1; m >>= 1) {
        p0 += __shfl_xor(p0, m, 32); p1 += __shfl_xor(p1, m, 32);
        p2 += __shfl_xor(p2, m, 32); p3 += __shfl_xor(p3, m, 32);
    }
    if (lane == 0) {
        float lg[4] = {p0, p1, p2, p3};
        int e1 = 0;
        for (int e = 1; e < 4; e++) if (lg[e] > lg[e1]) e1 = e;
        int e2 = -1;
        for (int e = 0; e < 4; e++) { if (e == e1) continue; if (e2 < 0 || lg[e] > lg[e2]) e2 = e; }
        float eb = expf(lg[e2] - lg[e1]);
        float g1 = 1.f / (1.f + eb), g2 = eb / (1.f + eb);
        float gv[4] = {0.f, 0.f, 0.f, 0.f};
        gv[e1] = g1; gv[e2] = g2;
        float* gr = gates + wave * 4;
        gr[0] = gv[0]; gr[1] = gv[1]; gr[2] = gv[2]; gr[3] = gv[3];
        atomicAdd(&imp[e1], g1);   atomicAdd(&imp[e2], g2);
        atomicAdd(&loadv[e1], 1.f); atomicAdd(&loadv[e2], 1.f);
    }
}

// ---------------------------------------------------------------------------
// Expert MLP via WMMA. One wave = TWO 16-edge tiles (M=32) of one expert.
// z[e][edge] = relu(feats @ W1[e] + b1[e]) @ W2[e] + b2[e]
// feats = [x[row] | x[col] | edge_attr]; K = 256 via 8x v_wmma_f32_16x16x32_f16
// per tile; edge_attr row (K index 256) folded fp32-exact into C init.
// B fragments come pre-swizzled from W1h: one coalesced 32 B/lane load, no LDS.
__global__ __launch_bounds__(128) void k_expert(
    const _Float16* __restrict__ xh, const _Float16* __restrict__ W1h,
    const float* __restrict__ W1, const float* __restrict__ b1,
    const float* __restrict__ W2, const float* __restrict__ b2,
    const int* __restrict__ rows, const int* __restrict__ cols,
    const float* __restrict__ ea, float* __restrict__ z,
    float* __restrict__ sumsq) {

    const int e     = blockIdx.y;
    const int wave  = threadIdx.x >> 5;
    const int lane  = threadIdx.x & 31;
    const int hh    = lane >> 4;                    // lane half (0/1)
    const int l16   = lane & 15;
    const int wtask = blockIdx.x * 4 + wave;        // < 10000
    const int base  = wtask * 32;                   // 32 consecutive edges

    // nodes for the two A tiles
    const int rn0 = rows[base + l16],      cn0 = cols[base + l16];
    const int rn1 = rows[base + 16 + l16], cn1 = cols[base + 16 + l16];

    // edge_attr for this lane's C rows: m = r + hh*8 (per tile)
    float ear0[8], ear1[8];
    #pragma unroll
    for (int r = 0; r < 8; r++) {
        ear0[r] = ea[base + r + hh * 8];
        ear1[r] = ea[base + 16 + r + hh * 8];
    }

    // A fragments (16x32 f16 per step), resident for whole task.
    // A layout: m = lane%16; halves 0..7 -> k = hh*8 + 0..7, halves 8..15 -> +16.
    v16h A0[8], A1[8];
    #pragma unroll
    for (int s = 0; s < 8; s++) {
        int fb = (s & 3) * 32 + hh * 8;
        {
            int node = (s < 4) ? rn0 : cn0;
            const v8h* px = (const v8h*)(xh + node * IND + fb);
            v8h g0 = px[0], g1 = px[2];
            A0[s] = __builtin_shufflevector(g0, g1, 0,1,2,3,4,5,6,7,8,9,10,11,12,13,14,15);
        }
        {
            int node = (s < 4) ? rn1 : cn1;
            const v8h* px = (const v8h*)(xh + node * IND + fb);
            v8h g0 = px[0], g1 = px[2];
            A1[s] = __builtin_shufflevector(g0, g1, 0,1,2,3,4,5,6,7,8,9,10,11,12,13,14,15);
        }
    }

    const float* W1last = W1 + e * FANIN * HID + 2 * IND * HID;  // K row 256 (fp32)
    const _Float16* Bbase = W1h + (size_t)e * 16 * 8 * 512;      // expert's fragments

    float zacc0[8], zacc1[8];
    #pragma unroll
    for (int r = 0; r < 8; r++) { zacc0[r] = 0.f; zacc1[r] = 0.f; }

    for (int ht = 0; ht < 16; ht++) {
        const int hc0 = ht * 16;
        const float b1v = b1[e * HID + hc0 + l16];
        const float wlv = W1last[hc0 + l16];
        const float w2v = W2[e * HID + hc0 + l16];
        const _Float16* Bht = Bbase + (size_t)ht * 8 * 512;

        if (ht < 15)   // prefetch next hid-tile's fragment block
            __builtin_prefetch(Bht + 8 * 512 + lane * 16, 0, 1);

        // C init: preact(m, n) = b1[n] + ea[m]*W1[256][n]   (m = r + hh*8, n = l16)
        v8f C0, C1;
        #pragma unroll
        for (int r = 0; r < 8; r++) {
            C0[r] = b1v + ear0[r] * wlv;
            C1[r] = b1v + ear1[r] * wlv;
        }

        #pragma unroll
        for (int ks = 0; ks < 8; ks++) {
            // B layout: n = lane%16; halves 0..7 -> k = hh*8 + 0..7, halves 8..15 -> +16
            v16h B = *(const v16h*)(Bht + (size_t)ks * 512 + lane * 16);
            C0 = __builtin_amdgcn_wmma_f32_16x16x32_f16(
                     false, A0[ks], false, B, (short)0, C0, false, false);
            C1 = __builtin_amdgcn_wmma_f32_16x16x32_f16(
                     false, A1[ks], false, B, (short)0, C1, false, false);
        }

        // relu then fold W2 column slice into per-lane partial of z
        #pragma unroll
        for (int r = 0; r < 8; r++) {
            float h0 = C0[r] > 0.f ? C0[r] : 0.f;
            float h1 = C1[r] > 0.f ? C1[r] : 0.f;
            zacc0[r] += h0 * w2v;
            zacc1[r] += h1 * w2v;
        }
    }

    // reduce across the 16 n-lanes within each lane-half
    #pragma unroll
    for (int r = 0; r < 8; r++) {
        #pragma unroll
        for (int m = 8; m >= 1; m >>= 1) {
            zacc0[r] += __shfl_xor(zacc0[r], m, 32);
            zacc1[r] += __shfl_xor(zacc1[r], m, 32);
        }
    }

    if (l16 == 0) {
        const float b2v = b2[e];
        float ss = 0.f;
        #pragma unroll
        for (int r = 0; r < 8; r++) {
            float z0 = zacc0[r] + b2v;
            float z1 = zacc1[r] + b2v;
            z[e * EE + base + hh * 8 + r]      = z0;
            z[e * EE + base + 16 + hh * 8 + r] = z1;
            ss += z0 * z0 + z1 * z1;
        }
        atomicAdd(&sumsq[e], ss);
    }
}

// ---------------------------------------------------------------------------
__global__ void k_rnorm(const float* sumsq, float* rnorm) {
    int e = threadIdx.x;
    if (e < NEXP) rnorm[e] = 1.f / fmaxf(sqrtf(sumsq[e]), 1e-12f);
}

// One wave per node: normalize, row-softmax over DEG=32, kth-largest threshold
// mask, gate-weighted mean over experts.
__global__ void k_output(const float* __restrict__ z, const float* __restrict__ gates,
                         const float* __restrict__ rnorm, float* __restrict__ out) {
    int wave = (blockIdx.x * blockDim.x + threadIdx.x) >> 5;
    int lane = threadIdx.x & 31;
    if (wave >= NN) return;
    const int KK[4] = {8, 16, 24, 29};   // max(1, round(32 * [.25,.5,.75,.9]))
    float acc = 0.f;
    #pragma unroll
    for (int e = 0; e < NEXP; e++) {
        float g  = gates[wave * 4 + e];
        float zv = z[e * EE + wave * DEG + lane] * rnorm[e];
        float mx = zv;
        #pragma unroll
        for (int m = 16; m >= 1; m >>= 1) mx = fmaxf(mx, __shfl_xor(mx, m, 32));
        float p = expf(zv - mx);
        float s = p;
        #pragma unroll
        for (int m = 16; m >= 1; m >>= 1) s += __shfl_xor(s, m, 32);
        float pi = p / s;
        // stable descending rank (ties broken by lane index) -> unique ranks
        int rk = 0;
        for (int j = 0; j < 32; j++) {
            float pj = __shfl(pi, j, 32);
            rk += (pj > pi) || (pj == pi && j < lane);
        }
        unsigned long long bal = __ballot(rk == KK[e] - 1);
        int src = __ffsll(bal) - 1;
        float thre = __shfl(pi, src, 32);
        float mask = (pi - thre + 1e-15f) > 0.f ? 1.f : 0.f;
        acc += g * mask * pi;
    }
    out[wave * DEG + lane] = acc * 0.25f;   // mean over NEXP=4
}

// ---------------------------------------------------------------------------
__device__ __forceinline__ float cv_sq(const float* v) {
    float mean = (v[0] + v[1] + v[2] + v[3]) * 0.25f;
    float var = 0.f;
    for (int i = 0; i < 4; i++) { float d = v[i] - mean; var += d * d; }
    var *= (1.f / 3.f);                    // ddof = 1
    return var / (mean * mean + 1e-10f);
}

__global__ void k_loss(const float* imp, const float* loadv, float* out) {
    if (threadIdx.x == 0 && blockIdx.x == 0) {
        float ci[4] = {imp[0], imp[1], imp[2], imp[3]};
        float cl[4] = {loadv[0], loadv[1], loadv[2], loadv[3]};
        out[EE] = 0.01f * (cv_sq(ci) + cv_sq(cl));
    }
}

// ---------------------------------------------------------------------------
extern "C" void kernel_launch(void* const* d_in, const int* in_sizes, int n_in,
                              void* d_out, int out_size, void* d_ws, size_t ws_size,
                              hipStream_t stream) {
    const float* x      = (const float*)d_in[0];
    const int*   eidx   = (const int*)  d_in[1];   // [2, E]
    const float* eattr  = (const float*)d_in[2];
    /* d_in[3] = temp (unused on eval path), d_in[5] = w_noise (unused) */
    const float* wg     = (const float*)d_in[4];
    const float* W1     = (const float*)d_in[6];
    const float* b1     = (const float*)d_in[7];
    const float* W2     = (const float*)d_in[8];
    const float* b2     = (const float*)d_in[9];
    float* out = (float*)d_out;

    char* ws = (char*)d_ws;
    float*    z     = (float*)   (ws + Z_OFF);
    float*    gates = (float*)   (ws + GATES_OFF);
    float*    sumsq = (float*)   (ws + SUMSQ_OFF);
    float*    imp   = (float*)   (ws + IMP_OFF);
    float*    loadv = (float*)   (ws + LOAD_OFF);
    float*    rnorm = (float*)   (ws + RNORM_OFF);
    _Float16* xh    = (_Float16*)(ws + XH_OFF);
    _Float16* W1h   = (_Float16*)(ws + W1H_OFF);

    const int* rows = eidx;
    const int* cols = eidx + EE;

    k_cvt_x <<<(NN * IND + 255) / 256, 256, 0, stream>>>(x, xh, NN * IND);
    k_cvt_w1<<<(NEXP * 16 * 8 * 32 + 255) / 256, 256, 0, stream>>>(W1, W1h);
    k_zero  <<<1, 32, 0, stream>>>(sumsq, imp, loadv);
    k_gating<<<(NN + 3) / 4, 128, 0, stream>>>(x, wg, gates, imp, loadv);
    k_expert<<<dim3((EE / 32) / 4, NEXP), 128, 0, stream>>>(
        xh, W1h, W1, b1, W2, b2, rows, cols, eattr, z, sumsq);
    k_rnorm <<<1, 32, 0, stream>>>(sumsq, rnorm);
    k_output<<<(NN + 3) / 4, 128, 0, stream>>>(z, gates, rnorm, out);
    k_loss  <<<1, 32, 0, stream>>>(imp, loadv, out);
}
